// GCNEncoder_72988674228409
// MI455X (gfx1250) — matrix operand order repaired and checked by reading, not compiled, not run on hardware
//
#include <hip/hip_runtime.h>

// ---------------------------------------------------------------------------
// GCN encoder for MI455X (gfx1250): bf16 WMMA GEMMs + L2-resident float
// atomic scatter aggregation.  out = D^-1/2 (A+I) D^-1/2 (h W) + b
// Factorized: hs = (hW)*dinv[row]; acc = hs (self loop) + scatter(hs[src]);
//             out = acc*dinv[row] + b  (optional ReLU)
// ---------------------------------------------------------------------------

typedef __attribute__((ext_vector_type(16))) __bf16 v16bf;
typedef __attribute__((ext_vector_type(8)))  float  v8f;
typedef unsigned int u32x4 __attribute__((ext_vector_type(4)));

union Frag16 {
    v16bf v;
    u32x4 u[2];
};

static __device__ __forceinline__ unsigned short f2bf(float f) {
    union { float f; unsigned int u; } x; x.f = f;
    unsigned int u = x.u;
    u += 0x7FFFu + ((u >> 16) & 1u);   // round-to-nearest-even
    return (unsigned short)(u >> 16);
}

// ---------------------------------------------------------------------------
// Degree / normalization kernels
// ---------------------------------------------------------------------------
__global__ void k_init_deg(float* __restrict__ deg, int n) {
    int i = blockIdx.x * blockDim.x + threadIdx.x;
    if (i < n) deg[i] = 1.0f;                       // self loop
}

__global__ void k_count_deg(const int* __restrict__ dst, float* __restrict__ deg, int e) {
    int i = blockIdx.x * blockDim.x + threadIdx.x;
    if (i < e) atomicAdd(&deg[dst[i]], 1.0f);
}

__global__ void k_rsqrt_deg(float* __restrict__ deg, int n) {
    int i = blockIdx.x * blockDim.x + threadIdx.x;
    if (i < n) deg[i] = rsqrtf(deg[i]);
}

// ---------------------------------------------------------------------------
// GEMM (K fixed = 128):  hs[row,:] = (X[row,:] @ W) * dinv[row]; acc := hs
// Block: 256 threads = 8 waves, 64 rows x H cols.  bf16 WMMA, f32 accum.
// LDS: X tile 64x128 bf16 (16KB) + W^T Hx128 bf16 (<=32KB)
// H is a compile-time constant so accumulators stay in fixed VGPRs
// (runtime H previously forced v_movrels/v_movreld around every WMMA).
// ---------------------------------------------------------------------------
template <int H>
__global__ __launch_bounds__(256) void k_gemm_prep(
    const float* __restrict__ X, const float* __restrict__ W,
    const float* __restrict__ dinv,
    float* __restrict__ hs, float* __restrict__ acc,
    int n)
{
    __shared__ unsigned short sX[64 * 128];    // [row][k]
    __shared__ unsigned short sW[H * 128];     // transposed: [col][k]

    const int t    = threadIdx.x;
    const int row0 = blockIdx.x * 64;

    // stage W transposed as bf16 (coalesced read of row-major [128][H])
    constexpr int WTOT = 128 * H;
#pragma unroll 4
    for (int idx = t; idx < WTOT; idx += 256) {
        int k = idx / H, c = idx % H;
        sW[c * 128 + k] = f2bf(W[idx]);
    }
    // stage X tile as bf16 (zero-pad rows >= n)
#pragma unroll 4
    for (int idx = t; idx < 64 * 128; idx += 256) {
        int r = idx >> 7, c = idx & 127;
        int grow = row0 + r;
        float v = (grow < n) ? X[(size_t)grow * 128 + c] : 0.0f;
        sX[idx] = f2bf(v);
    }
    __syncthreads();

    const int w    = t >> 5;
    const int lane = t & 31;
    const int half = lane >> 4;      // lane half (0: lanes 0-15, 1: 16-31)
    const int ml   = lane & 15;      // row (A) / col (B) within tile
    const int rt   = w >> 1;         // row tile 0..3
    constexpr int CTPW = H >> 5;     // col tiles per wave (128->4, 64->2)
    const int ct0  = (w & 1) * CTPW;

    v8f accv[CTPW] = {};             // col-tiles of 16x16 f32, fixed VGPRs

#pragma unroll
    for (int kb = 0; kb < 4; ++kb) {           // K = 128 = 4 x 32
        // A fragment: documented 16-bit A 16x32 layout
        // lane half 0 -> K in {0..7, 16..23}; half 1 -> {8..15, 24..31}
        Frag16 a;
        const int aoff = (rt * 16 + ml) * 128 + kb * 32 + half * 8;
        a.u[0] = *(const u32x4*)&sX[aoff];
        a.u[1] = *(const u32x4*)&sX[aoff + 16];

#pragma unroll
        for (int c = 0; c < CTPW; ++c) {
            // B fragment: 32x16 (KxN); lane half 0 -> K {0..7, 16..23}, half 1 -> {8..15, 24..31}
            Frag16 b;
            const int col  = (ct0 + c) * 16 + ml;
            const int boff = col * 128 + kb * 32 + half * 8;
            b.u[0] = *(const u32x4*)&sW[boff];
            b.u[1] = *(const u32x4*)&sW[boff + 16];

            accv[c] = __builtin_amdgcn_wmma_f32_16x16x32_bf16(
                false, a.v, false, b.v, (short)0, accv[c], false, false);
        }
    }

    // epilogue: scale by dinv[row], write hs and seed acc (self-loop term)
#pragma unroll
    for (int c = 0; c < CTPW; ++c) {
        const int gcol = (ct0 + c) * 16 + ml;
#pragma unroll
        for (int r = 0; r < 8; ++r) {
            const int grow = row0 + rt * 16 + r + half * 8;   // C/D: M = r + 8*half
            if (grow < n) {
                const float v = accv[c][r] * dinv[grow];
                const size_t o = (size_t)grow * H + gcol;
                hs[o]  = v;
                acc[o] = v;
            }
        }
    }
}

// ---------------------------------------------------------------------------
// Edge scatter: one wave per edge; lane covers H/32 contiguous floats.
// acc[dst,:] += hs[src,:]   (global_atomic_add_f32, L2-resident accumulator)
// ---------------------------------------------------------------------------
template <int H>
__global__ __launch_bounds__(256) void k_scatter(
    const int* __restrict__ srcs, const int* __restrict__ dsts,
    const float* __restrict__ hs, float* __restrict__ acc,
    int e)
{
    const int gw   = (blockIdx.x * blockDim.x + threadIdx.x) >> 5;
    const int lane = threadIdx.x & 31;
    if (gw >= e) return;
    const int s = srcs[gw];
    const int d = dsts[gw];
    if constexpr (H == 128) {
        const float4 v = *(const float4*)&hs[(size_t)s * 128 + lane * 4];
        float* p = &acc[(size_t)d * 128 + lane * 4];
        atomicAdd(p + 0, v.x);
        atomicAdd(p + 1, v.y);
        atomicAdd(p + 2, v.z);
        atomicAdd(p + 3, v.w);
    } else {  // H == 64
        const float2 v = *(const float2*)&hs[(size_t)s * 64 + lane * 2];
        float* p = &acc[(size_t)d * 64 + lane * 2];
        atomicAdd(p + 0, v.x);
        atomicAdd(p + 1, v.y);
    }
}

// ---------------------------------------------------------------------------
// Finalize: out = acc * dinv[row] + b[col]  (optional ReLU)
// ---------------------------------------------------------------------------
template <int H, bool RELU>
__global__ void k_finalize(const float* __restrict__ acc, const float* __restrict__ dinv,
                           const float* __restrict__ b, float* __restrict__ out,
                           int n)
{
    const size_t i = (size_t)blockIdx.x * blockDim.x + threadIdx.x;
    if (i >= (size_t)n * H) return;
    const int row = (int)(i / (size_t)H);
    const int col = (int)(i % (size_t)H);
    float v = acc[i] * dinv[row] + b[col];
    if (RELU) v = fmaxf(v, 0.0f);
    out[i] = v;
}

// ---------------------------------------------------------------------------
// Launch
// ---------------------------------------------------------------------------
extern "C" void kernel_launch(void* const* d_in, const int* in_sizes, int n_in,
                              void* d_out, int out_size, void* d_ws, size_t ws_size,
                              hipStream_t stream)
{
    const float* x   = (const float*)d_in[0];
    const int*   ei  = (const int*)  d_in[1];
    const float* W1  = (const float*)d_in[2];
    const float* b1  = (const float*)d_in[3];
    const float* W2  = (const float*)d_in[4];
    const float* b2  = (const float*)d_in[5];
    const float* Wmu = (const float*)d_in[6];
    const float* bmu = (const float*)d_in[7];
    const float* Wls = (const float*)d_in[8];
    const float* bls = (const float*)d_in[9];

    const int n = in_sizes[0] / 128;
    const int e = in_sizes[1] / 2;
    const int* src = ei;
    const int* dst = ei + e;

    float* ws   = (float*)d_ws;
    float* dinv = ws;                                 // [n]
    float* A    = ws + n;                             // hs  [n x 128]
    float* B    = A + (size_t)n * 128;                // acc [n x 128]
    float* C    = B + (size_t)n * 128;                // h   [n x 128]
    float* out  = (float*)d_out;

    const int tb = 256;
    // normalization
    k_init_deg <<<(n + tb - 1) / tb, tb, 0, stream>>>(dinv, n);
    k_count_deg<<<(e + tb - 1) / tb, tb, 0, stream>>>(dst, dinv, e);
    k_rsqrt_deg<<<(n + tb - 1) / tb, tb, 0, stream>>>(dinv, n);

    const int gblk = (n + 63) / 64;           // GEMM blocks (64 rows each)
    const int sblk = (e + 7) / 8;             // scatter: 8 waves (edges) / block

    // ---- layer 1: h1 = relu(conv(x, W1)) -> C
    k_gemm_prep<128><<<gblk, tb, 0, stream>>>(x, W1, dinv, A, B, n);
    k_scatter<128>  <<<sblk, tb, 0, stream>>>(src, dst, A, B, e);
    k_finalize<128, true><<<(unsigned)(((size_t)n * 128 + tb - 1) / tb), tb, 0, stream>>>(
        B, dinv, b1, C, n);

    // ---- layer 2: h2 = relu(conv(h1, W2)) -> C
    k_gemm_prep<128><<<gblk, tb, 0, stream>>>(C, W2, dinv, A, B, n);
    k_scatter<128>  <<<sblk, tb, 0, stream>>>(src, dst, A, B, e);
    k_finalize<128, true><<<(unsigned)(((size_t)n * 128 + tb - 1) / tb), tb, 0, stream>>>(
        B, dinv, b2, C, n);

    // ---- mu head: conv(h2, Wmu) -> out[0 : n*64]
    k_gemm_prep<64><<<gblk, tb, 0, stream>>>(C, Wmu, dinv, A, B, n);
    k_scatter<64>  <<<sblk, tb, 0, stream>>>(src, dst, A, B, e);
    k_finalize<64, false><<<(unsigned)(((size_t)n * 64 + tb - 1) / tb), tb, 0, stream>>>(
        B, dinv, bmu, out, n);

    // ---- logstd head: conv(h2, Wls) -> out[n*64 : 2*n*64]
    k_gemm_prep<64><<<gblk, tb, 0, stream>>>(C, Wls, dinv, A, B, n);
    k_scatter<64>  <<<sblk, tb, 0, stream>>>(src, dst, A, B, e);
    k_finalize<64, false><<<(unsigned)(((size_t)n * 64 + tb - 1) / tb), tb, 0, stream>>>(
        B, dinv, bls, out + (size_t)n * 64, n);
}
